// SwitchMLP_58858231824517
// MI455X (gfx1250) — matrix operand order repaired and checked by reading, not compiled, not run on hardware
//
#include <hip/hip_runtime.h>
#include <hip/hip_bf16.h>

#define T_TOK 8192
#define DIMD  1024
#define FFN_  4096
#define NEXP  8
#define XP    (DIMD + 8)   // LDS pitch for X tile (bf16 elems)
#define KC    64           // FFN chunk width
#define YP    (KC + 8)     // LDS pitch for Y chunk
#define MT_FB 16           // fallback M tile
#define MT_PK 32           // packed-path M tile

typedef __bf16 bf16_t;
typedef __attribute__((ext_vector_type(16))) __bf16        v16bf;
typedef __attribute__((ext_vector_type(8)))  float         v8f;
typedef __attribute__((ext_vector_type(4)))  unsigned int  u32x4;

union V16B { v16bf v; u32x4 q[2]; };

__device__ __forceinline__ bf16_t f2bf(float f) {
  union { float f; unsigned int u; } x; x.f = f;
  unsigned int r = x.u + 0x7FFFu + ((x.u >> 16) & 1u);   // round-to-nearest-even
  unsigned short h = (unsigned short)(r >> 16);
  return __builtin_bit_cast(bf16_t, h);
}

// ---------------- workspace layout (fast path) ----------------
#define WS_PROB_OFF   0ull
#define WS_CNT_OFF    0x10000ull                  // 64 KB
#define WS_TLIST_OFF  0x11000ull                  // 68 KB, E*T ints = 256 KB
#define WS_XB_OFF     0x100000ull                 // 1 MB, T*D bf16 = 16 MB
#define WS_PW1_OFF    (17ull << 20)               // 17 MB, 128 MB
#define WS_PW2_OFF    (145ull << 20)              // 145 MB, 64 MB
#define WS_NEED       (209ull << 20)

__global__ void moe_zero_counts(int* counts) {
  if (threadIdx.x < NEXP) counts[threadIdx.x] = 0;
}

__global__ void moe_router(const float* __restrict__ x,
                           const float* __restrict__ rw,
                           const float* __restrict__ rb,
                           float* __restrict__ prob,
                           int*   __restrict__ counts,
                           int*   __restrict__ tlist) {
  __shared__ float rwS[DIMD * NEXP];
  for (int i = threadIdx.x; i < DIMD * NEXP; i += blockDim.x) rwS[i] = rw[i];
  __syncthreads();

  int t = blockIdx.x * blockDim.x + threadIdx.x;
  if (t >= T_TOK) return;

  float acc[NEXP];
  #pragma unroll
  for (int e = 0; e < NEXP; ++e) acc[e] = rb[e];
  const float* xr = x + (size_t)t * DIMD;
  for (int d = 0; d < DIMD; ++d) {
    float xv = xr[d];
    #pragma unroll
    for (int e = 0; e < NEXP; ++e) acc[e] = fmaf(xv, rwS[d * NEXP + e], acc[e]);
  }
  float m = acc[0]; int am = 0;
  #pragma unroll
  for (int e = 1; e < NEXP; ++e) if (acc[e] > m) { m = acc[e]; am = e; }
  float s = 0.f;
  #pragma unroll
  for (int e = 0; e < NEXP; ++e) s += __expf(acc[e] - m);

  prob[t] = __builtin_amdgcn_rcpf(s);            // max_prob = 1/sum(exp(l-m))
  int pos = atomicAdd(&counts[am], 1);
  tlist[am * T_TOK + pos] = t;
}

// -------- pre-pack: x rows f32 -> bf16 (row major) --------
__global__ void cvt_x_bf16(const float* __restrict__ x, bf16_t* __restrict__ xb) {
  size_t base = ((size_t)blockIdx.x * 256 + threadIdx.x) * 8;
  const float4* p = (const float4*)(x + base);
  float4 f0 = p[0], f1 = p[1];
  union { u32x4 q; bf16_t h[8]; } o;
  o.h[0] = f2bf(f0.x); o.h[1] = f2bf(f0.y); o.h[2] = f2bf(f0.z); o.h[3] = f2bf(f0.w);
  o.h[4] = f2bf(f1.x); o.h[5] = f2bf(f1.y); o.h[6] = f2bf(f1.z); o.h[7] = f2bf(f1.w);
  *(u32x4*)(xb + base) = o.q;
}

// -------- pre-pack: weights f32 [E][K][N] -> bf16 WMMA-B tiles --------
// tile = 32(K) x 16(N); lane l (0-15): N=l, K=0..15 pair-packed; lanes 16-31:
// N=l-16, K=16..31. Each lane's 16 values contiguous: pw[e][nt][kt][lane*16+j]
__global__ void pack_weight(const float* __restrict__ w, bf16_t* __restrict__ pw,
                            int N, int ktiles) {
  const int lane = threadIdx.x;
  const int nt = blockIdx.x, kt = blockIdx.y, e = blockIdx.z;
  const int ntiles = N >> 4;
  const int n  = (nt << 4) + (lane & 15);
  const int kb = (kt << 5) + ((lane & 16) ? 16 : 0);
  const size_t K = (size_t)ktiles << 5;
  const float* src = w + (size_t)e * K * N + (size_t)kb * N + n;
  bf16_t* dst = pw + ((((size_t)e * ntiles + nt) * ktiles + kt) << 9) + lane * 16;
  V16B o;
  #pragma unroll
  for (int j = 0; j < 16; ++j) o.v[j] = f2bf(src[(size_t)j * N]);
  *(u32x4*)(dst)     = o.q[0];
  *(u32x4*)(dst + 8) = o.q[1];
}

// ================= fast path: packed bf16 GEMM, MT=32 =================
__launch_bounds__(256)
__global__ void moe_expert_ffn_packed(const bf16_t* __restrict__ xb,
                                      const bf16_t* __restrict__ pw1,
                                      const bf16_t* __restrict__ pw2,
                                      const int*   __restrict__ counts,
                                      const int*   __restrict__ tlist,
                                      const float* __restrict__ prob,
                                      float* __restrict__ out) {
  const int e   = blockIdx.y;
  const int cnt = counts[e];
  const int m0  = blockIdx.x * MT_PK;
  if (m0 >= cnt) return;
  const int rows = min(MT_PK, cnt - m0);

  __shared__ bf16_t Xs[MT_PK * XP];
  __shared__ float  As[MT_PK * KC];
  __shared__ float  Gs[MT_PK * KC];
  __shared__ bf16_t Ys[MT_PK * YP];
  __shared__ int    tkS[MT_PK];
  __shared__ float  prS[MT_PK];

  const int tid    = threadIdx.x;
  const int lane   = tid & 31;
  const int w      = tid >> 5;
  const int nlo    = lane & 15;
  const int hihalf = (lane >> 4) & 1;

  if (tid < MT_PK) {
    int tk = (tid < rows) ? tlist[e * T_TOK + m0 + tid] : -1;
    tkS[tid] = tk;
    prS[tid] = (tk >= 0) ? prob[tk] : 0.f;
  }
  __syncthreads();

  // ---- async-copy gathered bf16 X rows into LDS (ASYNCcnt path) ----
  {
    const unsigned xsbase = (unsigned)(uintptr_t)(void*)Xs;
    for (int rr = w; rr < MT_PK; rr += 8) {          // 4 rows per wave
      int tk = tkS[rr];                               // uniform per wave
      if (tk >= 0) {
        const char* gp = (const char*)(xb + (size_t)tk * DIMD);
        #pragma unroll
        for (int ch = 0; ch < 4; ++ch) {              // 4 x 512B per row
          unsigned      lb = xsbase + rr * (XP * 2) + ch * 512 + lane * 16;
          unsigned long long ga =
              (unsigned long long)(uintptr_t)(gp + ch * 512 + lane * 16);
          asm volatile("global_load_async_to_lds_b128 %0, %1, off"
                       :: "v"(lb), "v"(ga) : "memory");
        }
      } else {
        u32x4 z = {};
        #pragma unroll
        for (int ch = 0; ch < 4; ++ch)
          *(u32x4*)(&Xs[rr * XP + ch * 256 + lane * 8]) = z;
      }
    }
    asm volatile("s_wait_asynccnt 0x0" ::: "memory");
  }
  __syncthreads();

  const bf16_t* pw1e = pw1 + ((size_t)e << 23);  // 512 nt * 32 kt * 512
  const bf16_t* pw2e = pw2 + ((size_t)e << 22);  // 64 nt * 128 kt * 512

  v8f oacc[16];                                   // [m(2)][t(8)]
  #pragma unroll
  for (int t = 0; t < 16; ++t) oacc[t] = {};

  for (int fc = 0; fc < FFN_; fc += KC) {
    // ---------- stage 1: h = X @ w1 chunk (2 M-subtiles share B) ----------
    const int isG = (w >= 4);
    const int nt1 = (isG ? 256 : 0) + (fc >> 4) + (w & 3);
    const bf16_t* bq = pw1e + ((size_t)nt1 << 14) + lane * 16;
    __builtin_prefetch(pw2e + (((size_t)(w * 8) * 128 + (fc >> 5)) << 9) + lane * 16, 0, 1);

    v8f acc0 = {}, acc1 = {};
    for (int k0 = 0; k0 < DIMD; k0 += 32) {
      V16B A0, A1;
      const bf16_t* ap0 = &Xs[nlo * XP + k0 + hihalf * 16];
      const bf16_t* ap1 = ap0 + 16 * XP;
      A0.q[0] = *(const u32x4*)(ap0);
      A0.q[1] = *(const u32x4*)(ap0 + 8);
      A1.q[0] = *(const u32x4*)(ap1);
      A1.q[1] = *(const u32x4*)(ap1 + 8);
      V16B B;
      B.q[0] = *(const u32x4*)(bq);
      B.q[1] = *(const u32x4*)(bq + 8);
      bq += 512;
      acc0 = __builtin_amdgcn_wmma_f32_16x16x32_bf16(false, A0.v, false, B.v,
                                                     (short)0, acc0, false, false);
      acc1 = __builtin_amdgcn_wmma_f32_16x16x32_bf16(false, A1.v, false, B.v,
                                                     (short)0, acc1, false, false);
    }
    {
      float* dst = (isG ? Gs : As) + (w & 3) * 16;
      #pragma unroll
      for (int r = 0; r < 8; ++r) {
        dst[(r + hihalf * 8) * KC + nlo]        = acc0[r];
        dst[(16 + r + hihalf * 8) * KC + nlo]   = acc1[r];
      }
    }
    __syncthreads();

    // ---------- silu(a) * g -> bf16 ----------
    for (int i = tid; i < MT_PK * KC; i += 256) {
      int r = i >> 6, c = i & (KC - 1);
      float a = As[r * KC + c];
      float g = Gs[r * KC + c];
      float sig = __builtin_amdgcn_rcpf(1.0f + __expf(-a));
      Ys[r * YP + c] = f2bf(a * sig * g);
    }
    __syncthreads();

    // ---------- stage 2: out += Y @ w2 chunk ----------
    #pragma unroll
    for (int kk = 0; kk < KC; kk += 32) {
      V16B A0, A1;
      const bf16_t* ap0 = &Ys[nlo * YP + kk + hihalf * 16];
      const bf16_t* ap1 = ap0 + 16 * YP;
      A0.q[0] = *(const u32x4*)(ap0);
      A0.q[1] = *(const u32x4*)(ap0 + 8);
      A1.q[0] = *(const u32x4*)(ap1);
      A1.q[1] = *(const u32x4*)(ap1 + 8);

      const int kt2 = (fc + kk) >> 5;
      const bf16_t* b2 = pw2e + (((size_t)(w * 8) * 128 + kt2) << 9) + lane * 16;
      #pragma unroll
      for (int t = 0; t < 8; ++t) {
        V16B B;
        B.q[0] = *(const u32x4*)(b2);
        B.q[1] = *(const u32x4*)(b2 + 8);
        b2 += (size_t)128 << 9;                    // next N tile
        oacc[t]     = __builtin_amdgcn_wmma_f32_16x16x32_bf16(false, A0.v, false, B.v,
                                                              (short)0, oacc[t], false, false);
        oacc[8 + t] = __builtin_amdgcn_wmma_f32_16x16x32_bf16(false, A1.v, false, B.v,
                                                              (short)0, oacc[8 + t], false, false);
      }
    }
    __syncthreads();
  }

  #pragma unroll
  for (int m = 0; m < 2; ++m) {
    #pragma unroll
    for (int r = 0; r < 8; ++r) {
      int row = m * 16 + r + hihalf * 8;
      if (row < rows) {
        int tk  = tkS[row];
        float p = prS[row];
        float* orow = out + (size_t)tk * DIMD + w * 128 + nlo;
        #pragma unroll
        for (int t = 0; t < 8; ++t)
          orow[t * 16] = oacc[m * 8 + t][r] * p;
      }
    }
  }
}

// ================= fallback: convert-in-kernel GEMM (MT=16) =================
__launch_bounds__(256)
__global__ void moe_expert_ffn_fb(const float* __restrict__ x,
                                  const float* __restrict__ w1,
                                  const float* __restrict__ w2,
                                  const int*   __restrict__ counts,
                                  const int*   __restrict__ tlist,
                                  const float* __restrict__ prob,
                                  float* __restrict__ out) {
  const int e   = blockIdx.y;
  const int cnt = counts[e];
  const int m0  = blockIdx.x * MT_FB;
  if (m0 >= cnt) return;
  const int rows = min(MT_FB, cnt - m0);

  __shared__ bf16_t Xs[MT_FB * XP];
  __shared__ float  As[MT_FB * KC];
  __shared__ float  Gs[MT_FB * KC];
  __shared__ bf16_t Ys[MT_FB * YP];
  __shared__ int    tkS[MT_FB];
  __shared__ float  prS[MT_FB];

  const int tid    = threadIdx.x;
  const int lane   = tid & 31;
  const int w      = tid >> 5;
  const int nlo    = lane & 15;
  const int hihalf = (lane >> 4) & 1;

  if (tid < MT_FB) {
    int tk = (tid < rows) ? tlist[e * T_TOK + m0 + tid] : -1;
    tkS[tid] = tk;
    prS[tid] = (tk >= 0) ? prob[tk] : 0.f;
  }
  __syncthreads();

  for (int i = tid; i < MT_FB * DIMD; i += 256) {
    int r = i >> 10, c = i & (DIMD - 1);
    int tk = tkS[r];
    float v = (tk >= 0) ? x[(size_t)tk * DIMD + c] : 0.f;
    Xs[r * XP + c] = f2bf(v);
  }
  __syncthreads();

  const float* w1e = w1 + (size_t)e * DIMD * (2 * FFN_);
  const float* w2e = w2 + (size_t)e * FFN_ * DIMD;

  v8f oacc[8];
  #pragma unroll
  for (int t = 0; t < 8; ++t) oacc[t] = {};

  for (int fc = 0; fc < FFN_; fc += KC) {
    const int  isG     = (w >= 4);
    const int  colbase = (isG ? FFN_ : 0) + fc + (w & 3) * 16;
    const float* bp    = w1e + colbase + nlo;

    v8f acc = {};
    for (int k0 = 0; k0 < DIMD; k0 += 32) {
      V16B A;
      const bf16_t* ap = &Xs[nlo * XP + k0 + hihalf * 16];
      A.q[0] = *(const u32x4*)(ap);
      A.q[1] = *(const u32x4*)(ap + 8);
      V16B B;
      const float* bb = bp + (size_t)(k0 + hihalf * 16) * (2 * FFN_);
      #pragma unroll
      for (int j = 0; j < 8; ++j) {
        B.v[2 * j]     = f2bf(bb[(size_t)(2 * j)     * (2 * FFN_)]);
        B.v[2 * j + 1] = f2bf(bb[(size_t)(2 * j + 1) * (2 * FFN_)]);
      }
      acc = __builtin_amdgcn_wmma_f32_16x16x32_bf16(false, A.v, false, B.v,
                                                    (short)0, acc, false, false);
    }
    {
      float* dst = (isG ? Gs : As) + (w & 3) * 16;
      #pragma unroll
      for (int r = 0; r < 8; ++r)
        dst[(r + hihalf * 8) * KC + nlo] = acc[r];
    }
    __syncthreads();

    for (int i = tid; i < MT_FB * KC; i += 256) {
      int r = i >> 6, c = i & (KC - 1);
      float a = As[r * KC + c];
      float g = Gs[r * KC + c];
      float sig = __builtin_amdgcn_rcpf(1.0f + __expf(-a));
      Ys[r * YP + c] = f2bf(a * sig * g);
    }
    __syncthreads();

    #pragma unroll
    for (int kk = 0; kk < KC; kk += 32) {
      V16B A;
      const bf16_t* ap = &Ys[nlo * YP + kk + hihalf * 16];
      A.q[0] = *(const u32x4*)(ap);
      A.q[1] = *(const u32x4*)(ap + 8);

      const float* b2 = w2e + (size_t)(fc + kk + hihalf * 16) * DIMD + w * 128 + nlo;
      #pragma unroll
      for (int t = 0; t < 8; ++t) {
        const float* bb = b2 + t * 16;
        V16B B;
        #pragma unroll
        for (int j = 0; j < 8; ++j) {
          B.v[2 * j]     = f2bf(bb[(size_t)(2 * j)     * DIMD]);
          B.v[2 * j + 1] = f2bf(bb[(size_t)(2 * j + 1) * DIMD]);
        }
        oacc[t] = __builtin_amdgcn_wmma_f32_16x16x32_bf16(false, A.v, false, B.v,
                                                          (short)0, oacc[t], false, false);
      }
    }
    __syncthreads();
  }

  #pragma unroll
  for (int r = 0; r < 8; ++r) {
    int row = r + hihalf * 8;
    if (row < rows) {
      int tk  = tkS[row];
      float p = prS[row];
      float* orow = out + (size_t)tk * DIMD + w * 128 + nlo;
      #pragma unroll
      for (int t = 0; t < 8; ++t)
        orow[t * 16] = oacc[t][r] * p;
    }
  }
}

extern "C" void kernel_launch(void* const* d_in, const int* in_sizes, int n_in,
                              void* d_out, int out_size, void* d_ws, size_t ws_size,
                              hipStream_t stream) {
  const float* x  = (const float*)d_in[0];   // [T, D]
  const float* rw = (const float*)d_in[1];   // [D, E]
  const float* rb = (const float*)d_in[2];   // [E]
  const float* w1 = (const float*)d_in[3];   // [E, D, 2F]
  const float* w2 = (const float*)d_in[4];   // [E, F, D]
  float* out = (float*)d_out;                // [T, D]

  char*  ws     = (char*)d_ws;
  float* prob   = (float*)(ws + WS_PROB_OFF);
  int*   counts = (int*)(ws + WS_CNT_OFF);
  int*   tlist  = (int*)(ws + WS_TLIST_OFF);

  moe_zero_counts<<<1, 32, 0, stream>>>(counts);
  moe_router<<<T_TOK / 256, 256, 0, stream>>>(x, rw, rb, prob, counts, tlist);

  if (ws_size >= WS_NEED) {
    bf16_t* xb  = (bf16_t*)(ws + WS_XB_OFF);
    bf16_t* pw1 = (bf16_t*)(ws + WS_PW1_OFF);
    bf16_t* pw2 = (bf16_t*)(ws + WS_PW2_OFF);
    cvt_x_bf16<<<(T_TOK * DIMD) / (256 * 8), 256, 0, stream>>>(x, xb);
    pack_weight<<<dim3(512, 32, NEXP), 32, 0, stream>>>(w1, pw1, 2 * FFN_, DIMD / 32);
    pack_weight<<<dim3(64, 128, NEXP), 32, 0, stream>>>(w2, pw2, DIMD, FFN_ / 32);
    dim3 grid(T_TOK / MT_PK, NEXP, 1);
    moe_expert_ffn_packed<<<grid, 256, 0, stream>>>(xb, pw1, pw2, counts, tlist, prob, out);
  } else {
    dim3 grid(T_TOK / MT_FB, NEXP, 1);
    moe_expert_ffn_fb<<<grid, 256, 0, stream>>>(x, w1, w2, counts, tlist, prob, out);
  }
}
